// ConditionalPolice_17377437680145
// MI455X (gfx1250) — matrix operand order repaired and checked by reading, not compiled, not run on hardware
//
#include <hip/hip_runtime.h>
#include <hip/hip_bf16.h>
#include <math.h>

typedef __attribute__((ext_vector_type(16))) _Float16 v16h;
typedef __attribute__((ext_vector_type(8)))  float    v8f;

#define LEAKY 0.2f
#define EPSF  1e-16f

// ---------- helpers ----------
__device__ __forceinline__ unsigned fenc(float f) {
  unsigned b = __float_as_uint(f);
  return (b & 0x80000000u) ? ~b : (b | 0x80000000u);   // monotonic float->uint
}
__device__ __forceinline__ float fdec(unsigned u) {
  unsigned b = (u & 0x80000000u) ? (u ^ 0x80000000u) : ~u;
  return __uint_as_float(b);
}

// ---------- weight f32 -> f16 ----------
__global__ void cvt_f16_kernel(const float* __restrict__ in, _Float16* __restrict__ out, int n) {
  int i = blockIdx.x * blockDim.x + threadIdx.x;
  if (i < n) out[i] = (_Float16)in[i];
}

// ---------- WMMA node GEMM: H[M x DOUT] = X[M x 128] @ Wh[128 x DOUT] + b ----------
// One wave (32 lanes) per block computes one 16x16 tile. grid = (M/16, DOUT/16).
__global__ __launch_bounds__(32)
void gemm_node_wmma(const float* __restrict__ X, const _Float16* __restrict__ Wh,
                    const float* __restrict__ bias, float* __restrict__ H,
                    int M, int DOUT) {
  const int lane = threadIdx.x;            // 0..31 (wave32)
  const int row0 = blockIdx.x * 16;
  const int n0   = blockIdx.y * 16;
  const int m    = lane & 15;
  const int hi   = lane >> 4;              // 0/1 half-wave
  int r = row0 + m; if (r >= M) r = M - 1; // clamp (M is multiple of 16 here)
  const float* xrow = X + (size_t)r * 128;

  v8f c = {};
  for (int k0 = 0; k0 < 128; k0 += 32) {
    __builtin_prefetch(xrow + k0 + 32, 0, 0);
    v16h a, b;
    // A fragment (16x32 f16): lane holds M=lane&15; element i -> K offset below
    #pragma unroll
    for (int i = 0; i < 16; ++i) {
      int k = k0 + ((i >> 3) << 4) + (hi << 3) + (i & 7);
      a[i] = (_Float16)xrow[k];
    }
    // B fragment (32x16 f16): lane holds N = n0+(lane&15); element i -> same K offset as A
    #pragma unroll
    for (int i = 0; i < 16; ++i) {
      int k = k0 + ((i >> 3) << 4) + (hi << 3) + (i & 7);
      b[i] = Wh[(size_t)k * DOUT + n0 + m];
    }
    c = __builtin_amdgcn_wmma_f32_16x16x32_f16(false, a, false, b, (short)0, c, false, false);
  }
  // D fragment: VGPR v -> M = v + hi*8, N = lane&15
  #pragma unroll
  for (int v = 0; v < 8; ++v) {
    int mm = row0 + v + hi * 8;
    int nn = n0 + m;
    if (mm < M) H[(size_t)mm * DOUT + nn] = c[v] + bias[nn];
  }
}

// ---------- GEMV for DOUT==1 layer ----------
__global__ void gemv_node(const float* __restrict__ X, const float* __restrict__ w,
                          const float* __restrict__ b, float* __restrict__ out, int M) {
  int i = blockIdx.x * blockDim.x + threadIdx.x;
  if (i >= M) return;
  const float* xr = X + (size_t)i * 128;
  float s = 0.f;
  #pragma unroll 8
  for (int k = 0; k < 128; ++k) s = fmaf(xr[k], w[k], s);
  out[i] = s + b[0];
}

// ---------- edge pass A: scores + segment max (wave per edge, D in {32,128}) ----------
template <int D>
__global__ __launch_bounds__(256)
void edge_score(const int* __restrict__ ei, const float* __restrict__ ea,
                const float* __restrict__ H, const float* __restrict__ We,
                const float* __restrict__ att, float* __restrict__ scores,
                unsigned* __restrict__ smax, int E) {
  __shared__ float sWe[16 * D];
  __shared__ float sAtt[D];
  for (int t = threadIdx.x; t < 16 * D; t += blockDim.x) sWe[t] = We[t];
  for (int t = threadIdx.x; t < D; t += blockDim.x) sAtt[t] = att[t];
  __syncthreads();

  const int lane = threadIdx.x & 31, wid = threadIdx.x >> 5;
  const int e = blockIdx.x * 8 + wid;
  if (e >= E) return;
  const int src = ei[e], dst = ei[E + e];

  float eav = (lane < 16) ? ea[(size_t)e * 16 + lane] : 0.f;

  constexpr int CPL = D / 32;               // channels per lane
  float u[CPL];
  #pragma unroll
  for (int q = 0; q < CPL; ++q) {
    int ch = lane + 32 * q;
    u[q] = H[(size_t)src * D + ch] + H[(size_t)dst * D + ch];
  }
  #pragma unroll
  for (int d = 0; d < 16; ++d) {
    float ad = __shfl(eav, d, 32);          // broadcast edge attr d
    #pragma unroll
    for (int q = 0; q < CPL; ++q) u[q] = fmaf(ad, sWe[d * D + lane + 32 * q], u[q]);
  }
  float s = 0.f;
  #pragma unroll
  for (int q = 0; q < CPL; ++q) {
    float v = u[q];
    v = (v >= 0.f) ? v : LEAKY * v;         // leaky_relu
    s = fmaf(sAtt[lane + 32 * q], v, s);
  }
  #pragma unroll
  for (int off = 16; off; off >>= 1) s += __shfl_xor(s, off, 32);
  if (lane == 0) {
    scores[e] = s;
    atomicMax(&smax[dst], fenc(s));
  }
}

__global__ void edge_score_d1(const int* __restrict__ ei, const float* __restrict__ ea,
                              const float* __restrict__ H, const float* __restrict__ We,
                              const float* __restrict__ att, float* __restrict__ scores,
                              unsigned* __restrict__ smax, int E) {
  int e = blockIdx.x * blockDim.x + threadIdx.x;
  if (e >= E) return;
  int src = ei[e], dst = ei[E + e];
  float u = H[src] + H[dst];
  #pragma unroll
  for (int d = 0; d < 16; ++d) u = fmaf(ea[(size_t)e * 16 + d], We[d], u);
  u = (u >= 0.f) ? u : LEAKY * u;
  float s = att[0] * u;
  scores[e] = s;
  atomicMax(&smax[dst], fenc(s));
}

// ---------- edge pass B: out_i = sum(exp(s-m)*h_src) / (sum(exp)+eps) ----------
template <int D>
__global__ __launch_bounds__(256)
void edge_aggr(const int* __restrict__ ei, const float* __restrict__ H,
               const float* __restrict__ scores, const unsigned* __restrict__ smax,
               float* __restrict__ denom, float* __restrict__ numer, int E) {
  const int lane = threadIdx.x & 31, wid = threadIdx.x >> 5;
  const int e = blockIdx.x * 8 + wid;
  if (e >= E) return;
  const int src = ei[e], dst = ei[E + e];
  float ex = expf(scores[e] - fdec(smax[dst]));
  if (lane == 0) atomicAdd(&denom[dst], ex);
  constexpr int CPL = D / 32;
  #pragma unroll
  for (int q = 0; q < CPL; ++q) {
    int ch = lane + 32 * q;
    atomicAdd(&numer[(size_t)dst * D + ch], ex * H[(size_t)src * D + ch]);
  }
}

__global__ void edge_aggr_d1(const int* __restrict__ ei, const float* __restrict__ H,
                             const float* __restrict__ scores, const unsigned* __restrict__ smax,
                             float* __restrict__ denom, float* __restrict__ numer, int E) {
  int e = blockIdx.x * blockDim.x + threadIdx.x;
  if (e >= E) return;
  int src = ei[e], dst = ei[E + e];
  float ex = expf(scores[e] - fdec(smax[dst]));
  atomicAdd(&denom[dst], ex);
  atomicAdd(&numer[dst], ex * H[src]);
}

// ---------- finalize: out = numer/(denom+eps) + bias ----------
__global__ void finalize_nodes(const float* __restrict__ numer, const float* __restrict__ denom,
                               const float* __restrict__ bias, float* __restrict__ out,
                               int N, int D) {
  int idx = blockIdx.x * blockDim.x + threadIdx.x;
  if (idx >= N * D) return;
  int i = idx / D, c = idx - i * D;
  out[idx] = numer[idx] / (denom[i] + EPSF) + bias[c];
}

// ---------- threefry2x32-20 (JAX-compatible) + gumbel ----------
__device__ __forceinline__ unsigned rotl32(unsigned x, int r) { return (x << r) | (x >> (32 - r)); }

__device__ void threefry2x32(unsigned k0, unsigned k1, unsigned c0, unsigned c1,
                             unsigned* o0, unsigned* o1) {
  unsigned ks0 = k0, ks1 = k1, ks2 = 0x1BD11BDAu ^ k0 ^ k1;
  unsigned x0 = c0 + ks0, x1 = c1 + ks1;
  const int R0[4] = {13, 15, 26, 6}, R1[4] = {17, 29, 16, 24};
  unsigned ks[3] = {ks0, ks1, ks2};
  #pragma unroll
  for (int g = 0; g < 5; ++g) {
    const int* R = (g & 1) ? R1 : R0;
    #pragma unroll
    for (int i = 0; i < 4; ++i) { x0 += x1; x1 = rotl32(x1, R[i]); x1 ^= x0; }
    x0 += ks[(g + 1) % 3];
    x1 += ks[(g + 2) % 3] + (unsigned)(g + 1);
  }
  *o0 = x0; *o1 = x1;
}

__device__ __forceinline__ float bits_to_gumbel(unsigned bits) {
  const float tiny = 1.1754944e-38f;
  float f = __uint_as_float((bits >> 9) | 0x3F800000u) - 1.0f;  // [0,1)
  float u = f * (1.0f - tiny) + tiny;
  u = fmaxf(u, tiny);
  return -logf(-logf(u));
}

// ---------- final sampling (single block) ----------
__global__ __launch_bounds__(1024)
void sample_kernel(const float* __restrict__ nl, const float* __restrict__ al,
                   float* __restrict__ out, int N) {
  __shared__ float s_val[1024];
  __shared__ int   s_idx[1024];
  __shared__ float s_max[1024];
  const int tid = threadIdx.x;

  // key(42) = (0,42); split -> k1=(a0,a1), k2=(a2,a3) via counts [0,1,2,3]
  unsigned a0, a1, a2, a3;
  threefry2x32(0u, 42u, 0u, 2u, &a0, &a2);
  threefry2x32(0u, 42u, 1u, 3u, &a1, &a3);

  const int half = N >> 1;                  // N is even (50000)
  float bestv = -INFINITY; int besti = 0; float lmax = -INFINITY;
  for (int j = tid; j < half; j += 1024) {
    unsigned b0, b1;
    threefry2x32(a0, a1, (unsigned)j, (unsigned)(j + half), &b0, &b1);
    float l0 = nl[j], l1 = nl[j + half];
    float c0 = l0 + bits_to_gumbel(b0);
    float c1 = l1 + bits_to_gumbel(b1);
    if (c0 > bestv || (c0 == bestv && j < besti)) { bestv = c0; besti = j; }
    if (c1 > bestv || (c1 == bestv && (j + half) < besti)) { bestv = c1; besti = j + half; }
    lmax = fmaxf(lmax, fmaxf(l0, l1));
  }
  s_val[tid] = bestv; s_idx[tid] = besti; s_max[tid] = lmax;
  __syncthreads();
  for (int s = 512; s; s >>= 1) {
    if (tid < s) {
      float ov = s_val[tid + s]; int oi = s_idx[tid + s];
      if (ov > s_val[tid] || (ov == s_val[tid] && oi < s_idx[tid])) { s_val[tid] = ov; s_idx[tid] = oi; }
      s_max[tid] = fmaxf(s_max[tid], s_max[tid + s]);
    }
    __syncthreads();
  }
  const int nsel = s_idx[0];
  const float gmax = s_max[0];
  __syncthreads();

  float acc = 0.f;
  for (int j = tid; j < N; j += 1024) acc += expf(nl[j] - gmax);
  s_val[tid] = acc;
  __syncthreads();
  for (int s = 512; s; s >>= 1) { if (tid < s) s_val[tid] += s_val[tid + s]; __syncthreads(); }

  if (tid == 0) {
    float node_lp = nl[nsel] - gmax - logf(s_val[0]);
    const float* ar = al + (size_t)nsel * 32;
    float bv = -INFINITY; int bi = 0;
    for (int t = 0; t < 16; ++t) {
      unsigned b0, b1;
      threefry2x32(a2, a3, (unsigned)t, (unsigned)(t + 16), &b0, &b1);
      float c0 = ar[t] + bits_to_gumbel(b0);
      float c1 = ar[t + 16] + bits_to_gumbel(b1);
      if (c0 > bv || (c0 == bv && t < bi)) { bv = c0; bi = t; }
      if (c1 > bv || (c1 == bv && (t + 16) < bi)) { bv = c1; bi = t + 16; }
    }
    float am = -INFINITY;
    for (int t = 0; t < 32; ++t) am = fmaxf(am, ar[t]);
    float se = 0.f;
    for (int t = 0; t < 32; ++t) se += expf(ar[t] - am);
    float act_lp = ar[bi] - am - logf(se);
    out[0] = (float)nsel;
    out[1] = (float)bi;
    out[2] = node_lp + act_lp;
  }
}

// ================= host side =================
static inline size_t align256(size_t x) { return (x + 255) & ~(size_t)255; }

extern "C" void kernel_launch(void* const* d_in, const int* in_sizes, int n_in,
                              void* d_out, int out_size, void* d_ws, size_t ws_size,
                              hipStream_t stream) {
  const float* x    = (const float*)d_in[0];
  const int*   ei   = (const int*)  d_in[1];
  const float* ea   = (const float*)d_in[2];
  const float* Wl1  = (const float*)d_in[3];
  const float* bl1  = (const float*)d_in[4];
  const float* We1  = (const float*)d_in[5];
  const float* att1 = (const float*)d_in[6];
  const float* bias1= (const float*)d_in[7];
  const float* Wl2  = (const float*)d_in[8];
  const float* bl2  = (const float*)d_in[9];
  const float* We2  = (const float*)d_in[10];
  const float* att2 = (const float*)d_in[11];
  const float* bias2= (const float*)d_in[12];
  const float* Wl3  = (const float*)d_in[13];
  const float* bl3  = (const float*)d_in[14];
  const float* We3  = (const float*)d_in[15];
  const float* att3 = (const float*)d_in[16];
  const float* bias3= (const float*)d_in[17];
  float* out = (float*)d_out;

  const int N = in_sizes[0] / 128;
  const int E = in_sizes[1] / 2;

  // workspace carve-up
  char* p = (char*)d_ws;
  auto take = [&](size_t bytes) { char* r = p; p += align256(bytes); return r; };
  float*     h1      = (float*)    take((size_t)N * 128 * 4);
  float*     latent  = (float*)    take((size_t)N * 128 * 4);
  float*     h3      = (float*)    take((size_t)N * 32 * 4);
  float*     h2      = (float*)    take((size_t)N * 4);
  float*     scores  = (float*)    take((size_t)E * 4);
  unsigned*  smax    = (unsigned*) take((size_t)N * 4);
  float*     denom   = (float*)    take((size_t)N * 4);
  float*     numer   = (float*)    take((size_t)N * 128 * 4);
  float*     nlogit  = (float*)    take((size_t)N * 4);
  float*     alogit  = (float*)    take((size_t)N * 32 * 4);
  _Float16*  Wl1h    = (_Float16*) take(128 * 128 * 2);
  _Float16*  Wl3h    = (_Float16*) take(128 * 32 * 2);

  const int TPB = 256;
  const int rowTiles = (N + 15) / 16;
  const int edgeBlk8 = (E + 7) / 8;

  // convert weights to f16 for WMMA
  cvt_f16_kernel<<<(128 * 128 + TPB - 1) / TPB, TPB, 0, stream>>>(Wl1, Wl1h, 128 * 128);
  cvt_f16_kernel<<<(128 * 32 + TPB - 1) / TPB, TPB, 0, stream>>>(Wl3, Wl3h, 128 * 32);

  // ---------------- layer 1: D=128 ----------------
  gemm_node_wmma<<<dim3(rowTiles, 8), 32, 0, stream>>>(x, Wl1h, bl1, h1, N, 128);
  hipMemsetAsync(smax,  0, (size_t)N * 4, stream);
  hipMemsetAsync(denom, 0, (size_t)N * 4, stream);
  hipMemsetAsync(numer, 0, (size_t)N * 128 * 4, stream);
  edge_score<128><<<edgeBlk8, TPB, 0, stream>>>(ei, ea, h1, We1, att1, scores, smax, E);
  edge_aggr<128><<<edgeBlk8, TPB, 0, stream>>>(ei, h1, scores, smax, denom, numer, E);
  finalize_nodes<<<((size_t)N * 128 + TPB - 1) / TPB, TPB, 0, stream>>>(numer, denom, bias1, latent, N, 128);

  // ---------------- layer 2: D=1 (node logits) ----------------
  gemv_node<<<(N + TPB - 1) / TPB, TPB, 0, stream>>>(latent, Wl2, bl2, h2, N);
  hipMemsetAsync(smax,  0, (size_t)N * 4, stream);
  hipMemsetAsync(denom, 0, (size_t)N * 4, stream);
  hipMemsetAsync(numer, 0, (size_t)N * 4, stream);
  edge_score_d1<<<(E + TPB - 1) / TPB, TPB, 0, stream>>>(ei, ea, h2, We2, att2, scores, smax, E);
  edge_aggr_d1<<<(E + TPB - 1) / TPB, TPB, 0, stream>>>(ei, h2, scores, smax, denom, numer, E);
  finalize_nodes<<<(N + TPB - 1) / TPB, TPB, 0, stream>>>(numer, denom, bias2, nlogit, N, 1);

  // ---------------- layer 3: D=32 (action logits) ----------------
  gemm_node_wmma<<<dim3(rowTiles, 2), 32, 0, stream>>>(latent, Wl3h, bl3, h3, N, 32);
  hipMemsetAsync(smax,  0, (size_t)N * 4, stream);
  hipMemsetAsync(denom, 0, (size_t)N * 4, stream);
  hipMemsetAsync(numer, 0, (size_t)N * 32 * 4, stream);
  edge_score<32><<<edgeBlk8, TPB, 0, stream>>>(ei, ea, h3, We3, att3, scores, smax, E);
  edge_aggr<32><<<edgeBlk8, TPB, 0, stream>>>(ei, h3, scores, smax, denom, numer, E);
  finalize_nodes<<<((size_t)N * 32 + TPB - 1) / TPB, TPB, 0, stream>>>(numer, denom, bias3, alogit, N, 32);

  // ---------------- sampling ----------------
  sample_kernel<<<1, 1024, 0, stream>>>(nlogit, alogit, out, N);
}
